// LeafOnlyNet_14731737825597
// MI455X (gfx1250) — matrix operand order repaired and checked by reading, not compiled, not run on hardware
//
#include <hip/hip_runtime.h>
#include <hip/hip_bf16.h>

// Problem constants (match reference)
#define BB   4
#define NBB  256
#define LL   64
#define KVN  65          // K (kv length incl. block node)
#define CCH  512         // C
#define HH   8
#define HDN  64          // head dim
#define KVP  80          // kv padded to multiple of 16
#define KVP2 96          // kv padded to multiple of 32 (for attn @ V K-loop)
#define ATT_SCALE 0.125f // HD^-0.5

typedef __attribute__((ext_vector_type(16))) __bf16 v16bf;
typedef __attribute__((ext_vector_type(8)))  __bf16 v8bf;
typedef __attribute__((ext_vector_type(8)))  float  v8f;

union BF16x16 { v16bf v; v8bf h[2]; };

// ---- CDNA5 WMMA fragment loaders (wave32 layouts per ISA 7.12.2) ----
// A matrix 16x32 bf16, row-major source with leading dim ldm (bf16 units).
// Lane l: M = m0 + (l&15). lanes 0-15: VGPR0-3 K=0..7, VGPR4-7 K=16..23;
// lanes 16-31: VGPR0-3 K=8..15, VGPR4-7 K=24..31.
__device__ __forceinline__ v16bf loadA(const __bf16* p0, int ldm, int m0, int k0, int lane) {
    int m = m0 + (lane & 15), hi = lane >> 4;
    const __bf16* p = p0 + m * ldm + k0 + 8 * hi;
    BF16x16 u;
    u.h[0] = *(const v8bf*)(p);        // K = k0+8hi .. +7
    u.h[1] = *(const v8bf*)(p + 16);   // K = k0+16+8hi .. +7
    return u.v;
}
// B matrix 32x16 bf16 from "N-major" storage src[n][k] with leading dim ldn.
// Lane l: N = n0 + (l&15). lanes 0-15 hold K=0..7,16..23; lanes 16-31 hold K=8..15,24..31.
__device__ __forceinline__ v16bf loadB(const __bf16* p0, int ldn, int n0, int k0, int lane) {
    int n = n0 + (lane & 15), hi = lane >> 4;
    const __bf16* p = p0 + n * ldn + k0 + 8 * hi;
    BF16x16 u;
    u.h[0] = *(const v8bf*)(p);        // K = k0+8hi .. +7
    u.h[1] = *(const v8bf*)(p + 16);   // K = k0+16+8hi .. +7
    return u.v;
}

__device__ __forceinline__ v8f wmma_bf16(v16bf a, v16bf b, v8f c) {
    // (neg_a, A, neg_b, B, c_mod, C, reuse_a, reuse_b)
    return __builtin_amdgcn_wmma_f32_16x16x32_bf16(false, a, false, b, (short)0, c, false, false);
}

// ---- Weight prep: f32 row-major -> bf16 N-major (transposed) in workspace ----
__global__ __launch_bounds__(256) void prep_weights(const float* __restrict__ qkv_w,
                                                    const float* __restrict__ proj_w,
                                                    __bf16* __restrict__ wqkv_t,
                                                    __bf16* __restrict__ wproj_t) {
    int idx = blockIdx.x * 256 + threadIdx.x;
    const int NQ = 3 * CCH * CCH;   // 1536*512
    if (idx < NQ) {
        int n = idx / CCH, k = idx - n * CCH;            // wqkv_t[n][k]
        wqkv_t[idx] = (__bf16)qkv_w[k * (3 * CCH) + n];
    } else {
        int j = idx - NQ;
        if (j < CCH * CCH) {
            int n = j / CCH, k = j - n * CCH;            // wproj_t[n][k]
            wproj_t[j] = (__bf16)proj_w[k * CCH + n];
        }
    }
}

// ---- Fused block-attention kernel: one workgroup per (b, nb) ----
__global__ __launch_bounds__(256) void fused_block_attn(
    const float* __restrict__ x,        // (B, NB*L, C)
    const int*   __restrict__ amask,    // (NB, L, K) bool->int
    const float* __restrict__ ef,       // (NB, L, K, 4)
    const float* __restrict__ qkv_b,    // (3C)
    const float* __restrict__ proj_b,   // (C)
    const __bf16* __restrict__ wqkv_t,  // (3C, C) bf16 N-major
    const __bf16* __restrict__ wproj_t, // (C, C)  bf16 N-major
    float* __restrict__ out)            // (B, NB*L, C)
{
    extern __shared__ char smem[];
    const int LDKV = 520, LDQ = 72, LDK = 72, LDV = 96, LDSC = 84, LDA = 96, LDO = 520;
    __bf16* s_kv = (__bf16*)smem;                 // [80][520] kv block (bf16)
    __bf16* s_q  = s_kv + KVP * LDKV;             // [64][72]  q head
    __bf16* s_k  = s_q  + LL  * LDQ;              // [80][72]  k head (N-major for scores-B)
    __bf16* s_v  = s_k  + KVP * LDK;              // [64][96]  v head transposed: [hd][kv]
    __bf16* s_a  = s_v  + HDN * LDV;              // [64][96]  attn probs bf16
    __bf16* s_o  = s_a  + LL  * LDA;              // [64][520] attention output (bf16)
    float*  s_s  = (float*)(s_o + LL * LDO);      // [64][84]  scores f32

    const int tid = threadIdx.x, lane = tid & 31, wave = tid >> 5;
    const int g  = blockIdx.x;          // g = b*NB + nb
    const int nb = g & (NBB - 1);
    const float* xb = x + (size_t)g * (LL * CCH);

    // Phase 0: stage x block, compute block-node mean row, zero pad rows.
    for (int i = tid; i < LL * CCH; i += 256) {
        int r = i >> 9, c = i & (CCH - 1);
        s_kv[r * LDKV + c] = (__bf16)xb[r * CCH + c];
    }
    for (int c = tid; c < CCH; c += 256) {
        float s = 0.f;
        #pragma unroll 8
        for (int l = 0; l < LL; ++l) s += xb[l * CCH + c];
        s_kv[LL * LDKV + c] = (__bf16)(s * (1.0f / 64.0f));   // row 64 = mean
    }
    for (int i = tid; i < (KVP - KVN) * CCH; i += 256) {
        int r = KVN + i / CCH, c = i % CCH;
        s_kv[r * LDKV + c] = (__bf16)0.0f;
    }
    __syncthreads();

    for (int h = 0; h < HH; ++h) {
        // zero the K-pad region of s_v ([hd][80..95]) so attn-pad * v-pad == 0
        for (int i = tid; i < HDN * (KVP2 - KVP); i += 256) {
            int n = i >> 4, kk = KVP + (i & 15);
            s_v[n * LDV + kk] = (__bf16)0.0f;
        }
        // ---- per-head q/k/v GEMM: 16 + 20 + 20 tiles, K-loop 512/32 ----
        for (int t = wave; t < 56; t += 8) {
            int which, mt, nt;
            if (t < 16)      { which = 0; mt = t >> 2;          nt = t & 3; }
            else if (t < 36) { which = 1; mt = (t - 16) >> 2;   nt = (t - 16) & 3; }
            else             { which = 2; mt = (t - 36) >> 2;   nt = (t - 36) & 3; }
            int m0 = mt * 16, n0 = nt * 16;
            int ng = which * CCH + h * HDN + n0;        // column block in qkv_w
            const __bf16* wb = wqkv_t + (size_t)ng * CCH;
            v8f acc = {};
            for (int k0 = 0; k0 < CCH; k0 += 32) {
                if (k0 + 64 < CCH)
                    __builtin_prefetch((const void*)(wb + (lane & 15) * CCH + k0 + 64), 0, 1);
                v16bf a = loadA(s_kv, LDKV, m0, k0, lane);
                v16bf b = loadB(wb, CCH, 0, k0, lane);
                acc = wmma_bf16(a, b, acc);
            }
            int hi = lane >> 4, n = n0 + (lane & 15);
            float bias = qkv_b[ng + (lane & 15)];
            if (which == 0) {
                #pragma unroll
                for (int r = 0; r < 8; ++r)
                    s_q[(m0 + r + 8 * hi) * LDQ + n] = (__bf16)(acc[r] + bias);
            } else if (which == 1) {
                #pragma unroll
                for (int r = 0; r < 8; ++r)
                    s_k[(m0 + r + 8 * hi) * LDK + n] = (__bf16)(acc[r] + bias);
            } else {
                v8bf d;                                  // store V transposed: s_v[hd][kv]
                #pragma unroll
                for (int r = 0; r < 8; ++r) d[r] = (__bf16)(acc[r] + bias);
                *(v8bf*)(s_v + n * LDV + m0 + 8 * hi) = d;
            }
        }
        __syncthreads();

        // ---- scores = q @ k^T : 4x5 tiles, K-loop 64/32 ----
        for (int t = wave; t < 20; t += 8) {
            int m0 = (t / 5) * 16, n0 = (t % 5) * 16;
            v8f acc = {};
            for (int k0 = 0; k0 < HDN; k0 += 32) {
                v16bf a = loadA(s_q, LDQ, m0, k0, lane);
                v16bf b = loadB(s_k, LDK, n0, k0, lane);
                acc = wmma_bf16(a, b, acc);
            }
            int hi = lane >> 4, n = n0 + (lane & 15);
            #pragma unroll
            for (int r = 0; r < 8; ++r)
                s_s[(m0 + r + 8 * hi) * LDSC + n] = acc[r];
        }
        __syncthreads();

        // ---- softmax over K=65 with edge bias + mask (f32, one row/thread) ----
        if (tid < LL) {
            int l = tid;
            const float* efr = ef + (size_t)(nb * LL + l) * KVN * 4;
            const int*   mr  = amask + (size_t)(nb * LL + l) * KVN;
            float* sr = s_s + l * LDSC;
            float vmax = -1e30f;
            for (int kk = 0; kk < KVN; ++kk) {
                bool special = (kk == l) || (kk == KVN - 1);   // diag & block-node col
                float bias = special ? 1.0f : efr[kk * 4 + 3];
                bool ok = special || (mr[kk] != 0);
                float s = ok ? (sr[kk] * ATT_SCALE + bias) : -1e30f;
                sr[kk] = s;
                vmax = fmaxf(vmax, s);
            }
            float sum = 0.f;
            for (int kk = 0; kk < KVN; ++kk) {
                float e = __expf(sr[kk] - vmax);
                sr[kk] = e;
                sum += e;
            }
            float inv = 1.0f / sum;
            __bf16* ar = s_a + l * LDA;
            for (int kk = 0; kk < KVN; ++kk) ar[kk] = (__bf16)(sr[kk] * inv);
            for (int kk = KVN; kk < KVP2; ++kk) ar[kk] = (__bf16)0.0f;
        }
        __syncthreads();

        // ---- out_h = attn @ v : 4x4 tiles, K-loop 96/32 ----
        for (int t = wave; t < 16; t += 8) {
            int m0 = (t >> 2) * 16, n0 = (t & 3) * 16;
            v8f acc = {};
            for (int k0 = 0; k0 < KVP2; k0 += 32) {
                v16bf a = loadA(s_a, LDA, m0, k0, lane);
                v16bf b = loadB(s_v, LDV, n0, k0, lane);
                acc = wmma_bf16(a, b, acc);
            }
            int hi = lane >> 4, n = n0 + (lane & 15);
            #pragma unroll
            for (int r = 0; r < 8; ++r)
                s_o[(m0 + r + 8 * hi) * LDO + h * HDN + n] = (__bf16)acc[r];
        }
        __syncthreads();
    }

    // ---- proj: (64x512) @ (512x512) + b -> f32 out. 4x32 tiles, K-loop 512/32 ----
    float* ob = out + (size_t)g * (LL * CCH);
    for (int t = wave; t < 128; t += 8) {
        int m0 = (t >> 5) * 16, n0 = (t & 31) * 16;
        const __bf16* wb = wproj_t + (size_t)n0 * CCH;
        v8f acc = {};
        for (int k0 = 0; k0 < CCH; k0 += 32) {
            if (k0 + 64 < CCH)
                __builtin_prefetch((const void*)(wb + (lane & 15) * CCH + k0 + 64), 0, 1);
            v16bf a = loadA(s_o, LDO, m0, k0, lane);
            v16bf b = loadB(wb, CCH, 0, k0, lane);
            acc = wmma_bf16(a, b, acc);
        }
        int hi = lane >> 4, n = n0 + (lane & 15);
        float bias = proj_b[n];
        #pragma unroll
        for (int r = 0; r < 8; ++r)
            ob[(size_t)(m0 + r + 8 * hi) * CCH + n] = acc[r] + bias;
    }
}

extern "C" void kernel_launch(void* const* d_in, const int* in_sizes, int n_in,
                              void* d_out, int out_size, void* d_ws, size_t ws_size,
                              hipStream_t stream) {
    const float* x      = (const float*)d_in[0];
    const int*   amask  = (const int*)d_in[1];
    const float* ef     = (const float*)d_in[2];
    const float* qkv_w  = (const float*)d_in[3];
    const float* qkv_b  = (const float*)d_in[4];
    const float* proj_w = (const float*)d_in[5];
    const float* proj_b = (const float*)d_in[6];

    __bf16* wqkv_t  = (__bf16*)d_ws;                 // 1536*512 bf16 = 1.5 MB
    __bf16* wproj_t = wqkv_t + 3 * CCH * CCH;        // 512*512 bf16  = 0.5 MB

    // Weight transpose/convert (deterministic, redone every call; L2-resident)
    prep_weights<<<(3 * CCH * CCH + CCH * CCH) / 256, 256, 0, stream>>>(
        qkv_w, proj_w, wqkv_t, wproj_t);

    size_t smem = (size_t)(KVP * 520 + LL * 72 + KVP * 72 + HDN * 96 + LL * 96 + LL * 520) * 2
                + (size_t)LL * 84 * 4;               // ~216.6 KB of 320 KB/WGP
    fused_block_attn<<<BB * NBB, 256, smem, stream>>>(
        x, amask, ef, qkv_b, proj_b, wqkv_t, wproj_t, (float*)d_out);
}
